// AttentionBlock_171798692367
// MI455X (gfx1250) — matrix operand order repaired and checked by reading, not compiled, not run on hardware
//
#include <hip/hip_runtime.h>
#include <hip/hip_bf16.h>

typedef __attribute__((ext_vector_type(16))) _Float16 v16h;
typedef __attribute__((ext_vector_type(8)))  _Float16 v8h;
typedef __attribute__((ext_vector_type(8)))  float    v8f;

#define C_CH  256
#define NSP   4096
#define NHEAD 4
#define HD    64
#define NB    4

union AF { v16h v; v8h h2[2]; _Float16 e[16]; };

__device__ inline v8f wmma_f16(v16h a, v16h b, v8f c) {
  // 8 args: (neg_a, A, neg_b, B, c_mod, C, reuse_a, reuse_b)
  return __builtin_amdgcn_wmma_f32_16x16x32_f16(false, a, false, b, (short)0, c, false, false);
}

// Generic pointer to LDS byte offset (addrspace(3) offsets live in the low 32 bits).
__device__ inline unsigned lds_off(const void* p) { return (unsigned)(uintptr_t)p; }

// CDNA5 async global->LDS copy, 16B per lane, tracked by ASYNCcnt.
__device__ inline void async_b128(unsigned ldsoff, const void* g) {
  asm volatile("global_load_async_to_lds_b128 %0, %1, off"
               :: "v"(ldsoff), "v"(g)
               : "memory");
}

// ---------------- GroupNorm: x (b,c,h,w) fp32 -> xnT [b][n][c] f16 ----------------
__global__ void gn_kernel(const float* __restrict__ x, const float* __restrict__ w,
                          const float* __restrict__ bia, _Float16* __restrict__ xnT) {
  int b = blockIdx.x >> 5, g = blockIdx.x & 31;
  const float* xg = x + ((size_t)b * C_CH + g * 8) * NSP;
  float s = 0.f, sq = 0.f;
  for (int e = threadIdx.x; e < 8 * NSP; e += blockDim.x) {
    float v = xg[e];
    s += v; sq += v * v;
  }
  #pragma unroll
  for (int m = 16; m >= 1; m >>= 1) { s += __shfl_xor(s, m, 32); sq += __shfl_xor(sq, m, 32); }
  __shared__ float ss[8], sv[8];
  int wid = threadIdx.x >> 5;
  if ((threadIdx.x & 31) == 0) { ss[wid] = s; sv[wid] = sq; }
  __syncthreads();
  s = 0.f; sq = 0.f;
  for (int i = 0; i < (int)(blockDim.x >> 5); ++i) { s += ss[i]; sq += sv[i]; }
  const float inv = 1.0f / (8.0f * NSP);
  float mean = s * inv;
  float var  = sq * inv - mean * mean;
  float rstd = rsqrtf(var + 1e-5f);
  for (int e = threadIdx.x; e < 8 * NSP; e += blockDim.x) {
    int ci = e >> 12, n = e & (NSP - 1);
    int c = g * 8 + ci;
    float v = (xg[e] - mean) * rstd * w[c] + bia[c];
    xnT[((size_t)b * NSP + n) * C_CH + c] = (_Float16)v;
  }
}

// ---------------- QKV GEMM: (768x256) x (256x4096) per batch ----------------
// Writes Q,K as [b][h][n][d] f16, V as [b][h][d][n] f16.
__global__ void qkv_kernel(const _Float16* __restrict__ xnT,
                           const float* __restrict__ qkv_w, const float* __restrict__ qkv_b,
                           _Float16* __restrict__ qws, _Float16* __restrict__ kws,
                           _Float16* __restrict__ vws) {
  int lane = threadIdx.x & 31, wv = threadIdx.x >> 5;
  int hi = lane >> 4, lo16 = lane & 15;
  int b  = blockIdx.z;
  int m0 = blockIdx.y * 64 + wv * 16;   // output channel row base (o)
  int n0 = blockIdx.x * 64;             // spatial base
  const _Float16* Bbase = xnT + (size_t)b * NSP * C_CH;
  v8f acc[4] = {};
  for (int kc = 0; kc < C_CH; kc += 32) {
    AF a;
    const float* ar = qkv_w + (size_t)(m0 + lo16) * C_CH + kc + hi * 16;
    #pragma unroll
    for (int j = 0; j < 8; ++j) a.e[j]     = (_Float16)ar[j];
    #pragma unroll
    for (int j = 0; j < 8; ++j) a.e[8 + j] = (_Float16)ar[8 + j];
    #pragma unroll
    for (int t = 0; t < 4; ++t) {
      int n = n0 + t * 16 + lo16;
      v16h bf = *(const v16h*)(Bbase + (size_t)n * C_CH + kc + hi * 16);
      acc[t] = wmma_f16(a.v, bf, acc[t]);
    }
  }
  #pragma unroll
  for (int t = 0; t < 4; ++t) {
    #pragma unroll
    for (int r = 0; r < 8; ++r) {
      int o = m0 + r + 8 * hi;
      int n = n0 + t * 16 + lo16;
      _Float16 hv = (_Float16)(acc[t][r] + qkv_b[o]);
      if (o < 256) {
        int h = o >> 6, d = o & 63;
        qws[(((size_t)b * NHEAD + h) * NSP + n) * HD + d] = hv;
      } else if (o < 512) {
        int c = o - 256, h = c >> 6, d = c & 63;
        kws[(((size_t)b * NHEAD + h) * NSP + n) * HD + d] = hv;
      } else {
        int c = o - 512, h = c >> 6, d = c & 63;
        vws[(((size_t)b * NHEAD + h) * HD + d) * NSP + n] = hv;
      }
    }
  }
}

// ---------------- Flash attention ----------------
// 8 waves / block share async-staged K,V tiles in LDS (double buffered);
// each wave owns a 16-query tile.
__global__ void attn_kernel(const _Float16* __restrict__ qws, const _Float16* __restrict__ kws,
                            const _Float16* __restrict__ vws, _Float16* __restrict__ ao) {
  int tid = threadIdx.x;
  int lane = tid & 31, wv = tid >> 5;
  int hi = lane >> 4, lo16 = lane & 15;
  int b = blockIdx.z, h = blockIdx.y;
  int q0 = blockIdx.x * 128 + wv * 16;
  const _Float16* Q = qws + ((size_t)b * NHEAD + h) * NSP * HD;
  const _Float16* K = kws + ((size_t)b * NHEAD + h) * NSP * HD;
  const _Float16* V = vws + ((size_t)b * NHEAD + h) * HD * NSP;

  __shared__ __align__(16) _Float16 kbuf[2][32 * HD];   // [key][d]   4KB each
  __shared__ __align__(16) _Float16 vbuf[2][HD * 32];   // [d][key]   4KB each
  __shared__ __align__(16) _Float16 pbuf[8][16 * 32];   // per-wave P tile
  _Float16* pb = pbuf[wv];

  // Stage one 32-key chunk: 256 threads x (one 16B K copy + one 16B V copy).
  auto stage = [&](int chunk, int buf) {
    unsigned kl = lds_off(&kbuf[buf][0]);
    const _Float16* kg = K + (size_t)chunk * 32 * HD;   // contiguous 4KB
    async_b128(kl + tid * 16, kg + tid * 8);
    unsigned vl = lds_off(&vbuf[buf][0]);
    int d = tid >> 2, part = tid & 3;                   // 64 rows x 4 x 16B
    async_b128(vl + tid * 16, V + (size_t)d * NSP + chunk * 32 + part * 8);
  };

  // Q fragments for d=0..31 and d=32..63
  AF aq0, aq1;
  {
    const _Float16* qr = Q + (size_t)(q0 + lo16) * HD + hi * 16;
    aq0.h2[0] = *(const v8h*)(qr);
    aq0.h2[1] = *(const v8h*)(qr + 8);
    aq1.h2[0] = *(const v8h*)(qr + 32);
    aq1.h2[1] = *(const v8h*)(qr + 40);
  }
  v8f o0 = {}, o1 = {}, o2 = {}, o3 = {};
  float mrow[8], lrow[8];
  #pragma unroll
  for (int r = 0; r < 8; ++r) { mrow[r] = -3.0e38f; lrow[r] = 0.f; }
  const float scale = 0.125f;   // hd^-0.5, hd = 64
  const int NCH = NSP / 32;     // 128 chunks

  stage(0, 0);
  for (int ic = 0; ic < NCH; ++ic) {
    int cur = ic & 1;
    if (ic + 1 < NCH) {
      stage(ic + 1, cur ^ 1);
      // chunk ic's 2 copies are the oldest; async ops complete in order.
      asm volatile("s_wait_asynccnt 0x2" ::: "memory");
    } else {
      asm volatile("s_wait_asynccnt 0x0" ::: "memory");
    }
    __syncthreads();   // publish all waves' staged portions

    const _Float16* kb = kbuf[cur];
    const _Float16* vb = vbuf[cur];

    v8f s0 = {}, s1 = {};
    {
      const _Float16* kr0 = kb + (size_t)lo16 * HD + hi * 16;
      s0 = wmma_f16(aq0.v, *(const v16h*)(kr0),      s0);
      s0 = wmma_f16(aq1.v, *(const v16h*)(kr0 + 32), s0);
      const _Float16* kr1 = kb + (size_t)(16 + lo16) * HD + hi * 16;
      s1 = wmma_f16(aq0.v, *(const v16h*)(kr1),      s1);
      s1 = wmma_f16(aq1.v, *(const v16h*)(kr1 + 32), s1);
    }
    float corr[8];
    #pragma unroll
    for (int r = 0; r < 8; ++r) {
      float a  = s0[r] * scale, c2 = s1[r] * scale;
      float mx = fmaxf(a, c2);
      #pragma unroll
      for (int m = 1; m < 16; m <<= 1) mx = fmaxf(mx, __shfl_xor(mx, m, 32));
      float mnew = fmaxf(mrow[r], mx);
      float cf = __expf(mrow[r] - mnew);
      float p0 = __expf(a  - mnew);
      float p1 = __expf(c2 - mnew);
      float ps = p0 + p1;
      #pragma unroll
      for (int m = 1; m < 16; m <<= 1) ps += __shfl_xor(ps, m, 32);
      lrow[r] = lrow[r] * cf + ps;
      mrow[r] = mnew;
      corr[r] = cf;
      pb[(r + 8 * hi) * 32 + lo16]      = (_Float16)p0;
      pb[(r + 8 * hi) * 32 + 16 + lo16] = (_Float16)p1;
    }
    #pragma unroll
    for (int r = 0; r < 8; ++r) {
      o0[r] *= corr[r]; o1[r] *= corr[r]; o2[r] *= corr[r]; o3[r] *= corr[r];
    }
    AF ap;   // P tile 16x32 reloaded in A layout
    const _Float16* pr = pb + lo16 * 32 + hi * 16;
    ap.h2[0] = *(const v8h*)(pr);
    ap.h2[1] = *(const v8h*)(pr + 8);
    o0 = wmma_f16(ap.v, *(const v16h*)(vb + (size_t)(lo16)      * 32 + hi * 16), o0);
    o1 = wmma_f16(ap.v, *(const v16h*)(vb + (size_t)(16 + lo16) * 32 + hi * 16), o1);
    o2 = wmma_f16(ap.v, *(const v16h*)(vb + (size_t)(32 + lo16) * 32 + hi * 16), o2);
    o3 = wmma_f16(ap.v, *(const v16h*)(vb + (size_t)(48 + lo16) * 32 + hi * 16), o3);

    __syncthreads();   // all waves done reading before buffer is restaged
  }
  _Float16* aob = ao + (size_t)b * NSP * C_CH;
  #pragma unroll
  for (int r = 0; r < 8; ++r) {
    float invl = 1.0f / lrow[r];
    int q = q0 + r + 8 * hi;
    int cb = h * HD;
    aob[(size_t)q * C_CH + cb + lo16]      = (_Float16)(o0[r] * invl);
    aob[(size_t)q * C_CH + cb + 16 + lo16] = (_Float16)(o1[r] * invl);
    aob[(size_t)q * C_CH + cb + 32 + lo16] = (_Float16)(o2[r] * invl);
    aob[(size_t)q * C_CH + cb + 48 + lo16] = (_Float16)(o3[r] * invl);
  }
}

// ---------------- Proj GEMM + bias + residual -> fp32 out ----------------
__global__ void proj_kernel(const _Float16* __restrict__ ao, const float* __restrict__ pw,
                            const float* __restrict__ pbia, const float* __restrict__ x,
                            float* __restrict__ out) {
  int lane = threadIdx.x & 31, wv = threadIdx.x >> 5;
  int hi = lane >> 4, lo16 = lane & 15;
  int b  = blockIdx.z;
  int m0 = blockIdx.y * 64 + wv * 16;
  int n0 = blockIdx.x * 64;
  const _Float16* Bbase = ao + (size_t)b * NSP * C_CH;
  v8f acc[4] = {};
  for (int kc = 0; kc < C_CH; kc += 32) {
    AF a;
    const float* ar = pw + (size_t)(m0 + lo16) * C_CH + kc + hi * 16;
    #pragma unroll
    for (int j = 0; j < 8; ++j) a.e[j]     = (_Float16)ar[j];
    #pragma unroll
    for (int j = 0; j < 8; ++j) a.e[8 + j] = (_Float16)ar[8 + j];
    #pragma unroll
    for (int t = 0; t < 4; ++t) {
      int n = n0 + t * 16 + lo16;
      v16h bf = *(const v16h*)(Bbase + (size_t)n * C_CH + kc + hi * 16);
      acc[t] = wmma_f16(a.v, bf, acc[t]);
    }
  }
  #pragma unroll
  for (int t = 0; t < 4; ++t) {
    #pragma unroll
    for (int r = 0; r < 8; ++r) {
      int o = m0 + r + 8 * hi;
      int n = n0 + t * 16 + lo16;
      size_t idx = ((size_t)b * C_CH + o) * NSP + n;
      out[idx] = x[idx] + pbia[o] + acc[t][r];
    }
  }
}

extern "C" void kernel_launch(void* const* d_in, const int* in_sizes, int n_in,
                              void* d_out, int out_size, void* d_ws, size_t ws_size,
                              hipStream_t stream) {
  (void)in_sizes; (void)n_in; (void)out_size; (void)ws_size;
  const float* x      = (const float*)d_in[0];
  const float* gn_w   = (const float*)d_in[1];
  const float* gn_b   = (const float*)d_in[2];
  const float* qkv_w  = (const float*)d_in[3];
  const float* qkv_b  = (const float*)d_in[4];
  const float* proj_w = (const float*)d_in[5];
  const float* proj_b = (const float*)d_in[6];
  float* out = (float*)d_out;

  char* ws = (char*)d_ws;
  size_t seg = (size_t)NB * NSP * C_CH * sizeof(_Float16);   // 8 MB per buffer
  _Float16* xnT = (_Float16*)(ws);
  _Float16* qws = (_Float16*)(ws + 1 * seg);
  _Float16* kws = (_Float16*)(ws + 2 * seg);
  _Float16* vws = (_Float16*)(ws + 3 * seg);
  _Float16* ao  = (_Float16*)(ws + 4 * seg);

  gn_kernel  <<<dim3(NB * 32),              dim3(256), 0, stream>>>(x, gn_w, gn_b, xnT);
  qkv_kernel <<<dim3(NSP/64, 768/64, NB),   dim3(128), 0, stream>>>(xnT, qkv_w, qkv_b, qws, kws, vws);
  attn_kernel<<<dim3(NSP/128, NHEAD, NB),   dim3(256), 0, stream>>>(qws, kws, vws, ao);
  proj_kernel<<<dim3(NSP/64, C_CH/64, NB),  dim3(128), 0, stream>>>(ao, proj_w, proj_b, x, out);
}
